// GTP_2989297238263
// MI455X (gfx1250) — matrix operand order repaired and checked by reading, not compiled, not run on hardware
//
#include <hip/hip_runtime.h>
#include <hip/hip_bf16.h>
#include <math.h>

typedef __attribute__((ext_vector_type(16))) _Float16 v16h;
typedef __attribute__((ext_vector_type(8)))  _Float16 v8h;
typedef __attribute__((ext_vector_type(8)))  float    v8f;

static constexpr int   KH     = 32;      // hidden dim == WMMA K
static constexpr float BN_EPS = 1e-5f;

// ---------------------------------------------------------------------------
// A fragment: 16x32 f16, lane(r,half): K in {half*16..+15} for row r
// ---------------------------------------------------------------------------
__device__ __forceinline__ v16h load_a_frag(const _Float16* __restrict__ row, int half) {
    return *(const v16h*)(row + half * 16);
}
// B fragment: 32x16 f16, lane(r,half): K in {half*16..+15} for column r
__device__ __forceinline__ v16h load_b_frag(const _Float16* __restrict__ row, int half) {
    return *(const v16h*)(row + half * 16);
}

// ---------------------------------------------------------------------------
// Kernel 1: gather + BN(eval) + ReLU, fold proj_w into emb1 block.
// One wave (32 lanes) per node; lane = j index.
// ---------------------------------------------------------------------------
__global__ void gather_fold(const int* __restrict__ ids,
                            const float* __restrict__ emb1,
                            const float* __restrict__ emb2,
                            const float* __restrict__ g1g, const float* __restrict__ g1b,
                            const float* __restrict__ g1m, const float* __restrict__ g1v,
                            const float* __restrict__ g2g, const float* __restrict__ g2b,
                            const float* __restrict__ g2m, const float* __restrict__ g2v,
                            const float* __restrict__ pw,
                            _Float16* __restrict__ T16, _Float16* __restrict__ H216,
                            int total)
{
    const int wave = threadIdx.x >> 5;
    const int j    = threadIdx.x & 31;
    const int node = blockIdx.x * 8 + wave;
    if (node >= total) return;
    const int id = ids[node];
    const float* row = emb1 + (size_t)id * (KH * KH);
    float acc = 0.f;
#pragma unroll 4
    for (int k = 0; k < KH; k++) {
        const int e  = k * KH + j;                       // coalesced 128B per wave
        const float sc = g1g[e] * rsqrtf(g1v[e] + BN_EPS);
        float x = (row[e] - g1m[e]) * sc + g1b[e];
        acc = fmaf(pw[k], fmaxf(x, 0.f), acc);
    }
    T16[(size_t)node * KH + j] = (_Float16)acc;
    const float sc2 = g2g[j] * rsqrtf(g2v[j] + BN_EPS);
    float y = (emb2[(size_t)id * KH + j] - g2m[j]) * sc2 + g2b[j];
    H216[(size_t)node * KH + j] = (_Float16)fmaxf(y, 0.f);
}

// ---------------------------------------------------------------------------
// Kernel 2: online softmax stats (max, sum-exp) along the column axis of
// S = A x B^T (+bias, diag->0). Called with (T,H2) for row stats, (H2,T) for
// column stats (transposed view). One block = 16 rows; 8 waves split col-tiles.
// ---------------------------------------------------------------------------
__global__ void stats_kernel(const _Float16* __restrict__ A,
                             const _Float16* __restrict__ Bm,
                             const float* __restrict__ proj_b,
                             float* __restrict__ omax,
                             float* __restrict__ osum,
                             int N)
{
    const int b       = blockIdx.y;
    const int rowbase = blockIdx.x * 16;
    const int wave    = threadIdx.x >> 5;
    const int lane    = threadIdx.x & 31;
    const int r       = lane & 15;
    const int half    = lane >> 4;
    const int ntiles  = N >> 4;
    const float bias  = proj_b[0];

    const _Float16* Ab = A  + (size_t)b * N * KH;
    const _Float16* Bb = Bm + (size_t)b * N * KH;

    const v16h afrag = load_a_frag(Ab + (size_t)(rowbase + r) * KH, half);

    float mv[8], sv[8];
#pragma unroll
    for (int v = 0; v < 8; v++) { mv[v] = -__builtin_inff(); sv[v] = 0.f; }

    for (int ct = wave; ct < ntiles; ct += 8) {
        const int colbase = ct * 16;
        const v16h bfrag = load_b_frag(Bb + (size_t)(colbase + r) * KH, half);
        v8f c = {};
        c = __builtin_amdgcn_wmma_f32_16x16x32_f16(false, afrag, false, bfrag,
                                                   (short)0, c, false, false);
#pragma unroll
        for (int v = 0; v < 8; v++) {
            const int row_g = rowbase + v + 8 * half;
            const int col_g = colbase + r;
            const float s  = (row_g == col_g) ? 0.f : (c[v] + bias);
            const float Mn = fmaxf(mv[v], s);
            sv[v] = sv[v] * __expf(mv[v] - Mn) + __expf(s - Mn);
            mv[v] = Mn;
        }
    }

    // combine across the 16 lanes sharing a half (rows are fixed per half)
#pragma unroll
    for (int off = 1; off < 16; off <<= 1) {
#pragma unroll
        for (int v = 0; v < 8; v++) {
            const float om = __shfl_xor(mv[v], off, 32);
            const float os = __shfl_xor(sv[v], off, 32);
            const float Mn = fmaxf(mv[v], om);
            sv[v] = sv[v] * __expf(mv[v] - Mn) + os * __expf(om - Mn);
            mv[v] = Mn;
        }
    }

    // combine across the 8 waves via LDS
    __shared__ float lm[8][16];
    __shared__ float ls[8][16];
    if (r == 0) {
#pragma unroll
        for (int v = 0; v < 8; v++) {
            lm[wave][v + 8 * half] = mv[v];
            ls[wave][v + 8 * half] = sv[v];
        }
    }
    __syncthreads();
    if (threadIdx.x < 16) {
        float M = lm[0][threadIdx.x], S = ls[0][threadIdx.x];
#pragma unroll
        for (int w = 1; w < 8; w++) {
            const float m2 = lm[w][threadIdx.x], s2 = ls[w][threadIdx.x];
            const float Mn = fmaxf(M, m2);
            S = S * __expf(M - Mn) + s2 * __expf(m2 - Mn);
            M = Mn;
        }
        omax[(size_t)b * N + rowbase + threadIdx.x] = M;
        osum[(size_t)b * N + rowbase + threadIdx.x] = S;
    }
}

// ---------------------------------------------------------------------------
// Kernel 3: recompute S tile via WMMA, apply both softmaxes, stream outputs.
// grid = (N/128, N/16, B); block = 256 (8 waves, one 16x16 tile each).
// ---------------------------------------------------------------------------
__global__ void final_kernel(const _Float16* __restrict__ T16,
                             const _Float16* __restrict__ H216,
                             const float* __restrict__ rmax,
                             const float* __restrict__ rsum,
                             const float* __restrict__ cmax,
                             const float* __restrict__ csum,
                             const float* __restrict__ proj_b,
                             float* __restrict__ outm,
                             float* __restrict__ outmask,
                             int N)
{
    const int b       = blockIdx.z;
    const int rowbase = blockIdx.y * 16;
    const int wave    = threadIdx.x >> 5;
    const int ct      = blockIdx.x * 8 + wave;
    const int lane    = threadIdx.x & 31;
    const int r       = lane & 15;
    const int half    = lane >> 4;
    const int colbase = ct * 16;
    const float bias  = proj_b[0];

    const _Float16* Ab = T16  + (size_t)b * N * KH;
    const _Float16* Bb = H216 + (size_t)b * N * KH;
    const v16h a  = load_a_frag(Ab + (size_t)(rowbase + r) * KH, half);
    const v16h bf = load_b_frag(Bb + (size_t)(colbase + r) * KH, half);
    v8f c = {};
    c = __builtin_amdgcn_wmma_f32_16x16x32_f16(false, a, false, bf,
                                               (short)0, c, false, false);

    const float* rM = rmax + (size_t)b * N;
    const float* rS = rsum + (size_t)b * N;
    const float* cM = cmax + (size_t)b * N;
    const float* cS = csum + (size_t)b * N;
    const int   col_g = colbase + r;
    const float cm = cM[col_g];
    const float ci = 1.f / cS[col_g];
    const size_t base = ((size_t)b * N + rowbase) * N;
#pragma unroll
    for (int v = 0; v < 8; v++) {
        const int m_local = v + 8 * half;
        const int row_g   = rowbase + m_local;
        const float s   = (row_g == col_g) ? 0.f : (c[v] + bias);
        const float val = (__expf(s - rM[row_g]) / rS[row_g]) * (__expf(s - cm) * ci);
        const size_t idx = base + (size_t)m_local * N + col_g;
        outm[idx]    = val;
        outmask[idx] = (val > 1e-6f) ? 1.0f : 0.0f;
    }
}

// ---------------------------------------------------------------------------
extern "C" void kernel_launch(void* const* d_in, const int* in_sizes, int n_in,
                              void* d_out, int out_size, void* d_ws, size_t ws_size,
                              hipStream_t stream) {
    const int*   node_ids = (const int*)  d_in[0];
    const float* emb1     = (const float*)d_in[1];
    const float* emb2     = (const float*)d_in[2];
    const float* g1g      = (const float*)d_in[3];
    const float* g1b      = (const float*)d_in[4];
    const float* g1m      = (const float*)d_in[5];
    const float* g1v      = (const float*)d_in[6];
    const float* g2g      = (const float*)d_in[7];
    const float* g2b      = (const float*)d_in[8];
    const float* g2m      = (const float*)d_in[9];
    const float* g2v      = (const float*)d_in[10];
    const float* pw       = (const float*)d_in[11];
    const float* pb       = (const float*)d_in[12];

    const int BN = in_sizes[0];                               // B*N = 16384
    const int N  = (int)((long long)out_size / (2LL * BN));   // out = 2*B*N*N -> N
    const int B  = BN / N;

    // workspace layout (~2.4 MB)
    _Float16* T16  = (_Float16*)d_ws;                 // BN*32 f16
    _Float16* H216 = T16 + (size_t)BN * KH;           // BN*32 f16
    float* rmax = (float*)(H216 + (size_t)BN * KH);   // BN f32
    float* rsum = rmax + BN;
    float* cmax = rsum + BN;
    float* csum = cmax + BN;

    float* outm    = (float*)d_out;
    float* outmask = outm + (size_t)B * N * N;

    // 1) gather + BN + ReLU + fold proj_w  ->  T (B,N,32), H2 (B,N,32) in f16
    gather_fold<<<dim3((BN + 7) / 8), dim3(256), 0, stream>>>(
        node_ids, emb1, emb2, g1g, g1b, g1m, g1v, g2g, g2b, g2m, g2v, pw,
        T16, H216, BN);

    // 2) row stats: softmax over axis=2 (per (b,n))
    stats_kernel<<<dim3(N / 16, B), dim3(256), 0, stream>>>(T16, H216, pb, rmax, rsum, N);
    // 3) col stats: softmax over axis=1 (per (b,m)) via transposed product
    stats_kernel<<<dim3(N / 16, B), dim3(256), 0, stream>>>(H216, T16, pb, cmax, csum, N);

    // 4) recompute tiles + apply both softmaxes + write matrix & mask
    final_kernel<<<dim3(N / 128, N / 16, B), dim3(256), 0, stream>>>(
        T16, H216, rmax, rsum, cmax, csum, pb, outm, outmask, N);
}
